// CustomLSTM_44066364457117
// MI455X (gfx1250) — matrix operand order repaired and checked by reading, not compiled, not run on hardware
//
#include <hip/hip_runtime.h>

// ---- CDNA5 vector types ----
typedef __attribute__((ext_vector_type(16))) _Float16 v16h;
typedef __attribute__((ext_vector_type(8)))  _Float16 v8h;
typedef __attribute__((ext_vector_type(8)))  float    v8f;
typedef __attribute__((ext_vector_type(4)))  float    f4;

#define SEQ    4096
#define BATCH  512
#define INPUT  8
#define HIDDEN 30

__device__ __forceinline__ float fast_rcp(float x) {
#if __has_builtin(__builtin_amdgcn_rcpf)
    return __builtin_amdgcn_rcpf(x);
#else
    return 1.0f / x;
#endif
}
__device__ __forceinline__ float tanh_fast(float x) {
#if __has_builtin(__builtin_amdgcn_tanhf)
    return __builtin_amdgcn_tanhf(x);          // native v_tanh_f32 (confirmed in asm)
#else
    return 2.0f * fast_rcp(1.0f + __expf(-2.0f * x)) - 1.0f;
#endif
}
__device__ __forceinline__ float sigmoidf(float x) {
#if __has_builtin(__builtin_amdgcn_tanhf)
    // 1 trans + 2 VALU instead of 2 trans + 2 VALU
    return fmaf(0.5f, tanh_fast(0.5f * x), 0.5f);
#else
    return fast_rcp(1.0f + __expf(-x));
#endif
}

// One wave32 per block; one block per 16-batch tile; 32 blocks total.
// Gate columns are permuted: tile T (T=0..7): gate g = T&3, hidden group grp = T>>2,
// column within tile n = lane&15 -> hidden unit hh = grp*16 + n (valid iff hh < 30).
// Thus acc[grp*4+g][r] holds gate g of element (M = r + 8*half, hh = grp*16 + l16):
// the four gates of one element sit in the same lane/slot -> in-register cell update.
// Pad columns (hh>=30) carry zero weights in both B operands, and tanh/sigmoid keep h
// bounded, so pad values are finite and contribute exactly 0 to every WMMA/dot.
__global__ __launch_bounds__(32)
void lstm_wmma_kernel(const float* __restrict__ x,
                      const float* __restrict__ W_ih,
                      const float* __restrict__ W_hh,
                      const float* __restrict__ b_ih,
                      const float* __restrict__ b_hh,
                      const float* __restrict__ W_lin,
                      const float* __restrict__ b_lin,
                      float* __restrict__ out)
{
    const int lane = threadIdx.x;      // 0..31
    const int half = lane >> 4;        // 0 / 1
    const int l16  = lane & 15;
    const int b0   = blockIdx.x * 16;  // batch tile base

    // ---------- persistent weight operands in VGPRs (WMMA B-layout) ----------
    // B operand in this code's contiguous k order (32x16 KxN): lanes 0-15 hold K=0..15 (elem j -> k=j),
    // lanes 16-31 hold K=16..31 (elem j -> k=16+j); column n = l16.
    v16h B1[8];  // W_hh^T, rows k=0..29, permuted columns
    v16h B2[8];  // rows k=0..7 = W_ih^T, row k=8 = (b_ih+b_hh), rest 0
    #pragma unroll
    for (int T = 0; T < 8; ++T) {
        const int g   = T & 3;
        const int grp = T >> 2;
        const int hh  = grp * 16 + l16;
        const bool colv = (hh < HIDDEN);
        const int nco = g * HIDDEN + hh;          // original gate-major column
        v16h w1 = {}, w2 = {};
        #pragma unroll
        for (int j = 0; j < 16; ++j) {
            const int k = half * 16 + j;
            float v1 = 0.0f, v2 = 0.0f;
            if (colv) {
                if (k < HIDDEN) v1 = W_hh[nco * HIDDEN + k];
                if (k < INPUT)       v2 = W_ih[nco * INPUT + k];
                else if (k == INPUT) v2 = b_ih[nco] + b_hh[nco];
            }
            w1[j] = (_Float16)v1;
            w2[j] = (_Float16)v2;
        }
        B1[T] = w1;
        B2[T] = w2;
    }

    // W_lin slice for the fused output dot: this lane covers k = half*16 + j
    float wl[16];
    #pragma unroll
    for (int j = 0; j < 16; ++j) {
        const int k = half * 16 + j;
        wl[j] = (k < HIDDEN) ? W_lin[k] : 0.0f;
    }
    const float blin = b_lin[0];

    // ---------- LDS: h staged in WMMA A-layout row-major hbuf[M][k], 16x32 f16 ----------
    __shared__ __align__(16) _Float16 hbuf[16 * 32];
    for (int i = lane; i < 16 * 32; i += 32) hbuf[i] = (_Float16)0.0f;
    __syncthreads();
    const v8h* __restrict__ hp = (const v8h*)hbuf;

    // cell state c for (M = r + 8*half, hh = grp*16 + l16), index ci = grp*8 + r
    float cst[16];
    #pragma unroll
    for (int i = 0; i < 16; ++i) cst[i] = 0.0f;

    // ---------- prefetch x row for t = 0 ----------
    const float* xrow = x + ((size_t)b0 + l16) * INPUT;
    f4 xa = *(const f4*)(xrow);
    f4 xb = *(const f4*)(xrow + 4);

    const _Float16 hzero = (_Float16)0.0f;
    const _Float16 hone  = (_Float16)1.0f;

    for (int t = 0; t < SEQ; ++t) {
        // load h[t-1] in A-layout EARLY, straight into the contiguous WMMA operand
        // (two ds_load_b128 at byte offsets M*64 + half*32 and +16; latency covered
        //  by the x-projection WMMAs below)
        union { v16h v; v8h h[2]; } hU;
        hU.h[0] = hp[l16 * 4 + half * 2];
        hU.h[1] = hp[l16 * 4 + half * 2 + 1];

        // Build xA branchlessly (A-layout 16x32):
        // lanes 0-15:  elem 0..7 -> K0..7 = x[t,M,:],  elem 8 -> K8 = 1.0 (bias row)
        // lanes16-31:  all 0
        v16h xA = {};
        #pragma unroll
        for (int j = 0; j < 4; ++j) {
            xA[j]     = (half == 0) ? (_Float16)xa[j] : hzero;
            xA[4 + j] = (half == 0) ? (_Float16)xb[j] : hzero;
        }
        xA[8] = (half == 0) ? hone : hzero;

        // input-projection WMMAs (independent of h -> fill the WMMA pipe while DS loads land)
        v8f acc[8];
        const v8f zc = {};
        #pragma unroll
        for (int T = 0; T < 8; ++T)
            acc[T] = __builtin_amdgcn_wmma_f32_16x16x32_f16(
                         false, xA, false, B2[T], (short)0, zc, false, false);

        // prefetch x for t+1 (latency hidden behind this step's compute)
        if (t + 1 < SEQ) {
            const float* nrow = x + ((size_t)(t + 1) * BATCH + b0 + l16) * INPUT;
            xa = *(const f4*)(nrow);
            xb = *(const f4*)(nrow + 4);
        }

        // recurrent WMMAs: acc += h . W_hh^T   (8x v_wmma_f32_16x16x32_f16)
        #pragma unroll
        for (int T = 0; T < 8; ++T)
            acc[T] = __builtin_amdgcn_wmma_f32_16x16x32_f16(
                         false, hU.v, false, B1[T], (short)0, acc[T], false, false);

        // all hbuf reads for this step are done (hA above; dot reads come after store sync)
        __syncthreads();

        // ---------- in-register cell update + h store ----------
        #pragma unroll
        for (int grp = 0; grp < 2; ++grp) {
            const int hh = grp * 16 + l16;
            #pragma unroll
            for (int r = 0; r < 8; ++r) {
                const float ig = sigmoidf(acc[grp * 4 + 0][r]);
                const float fg = sigmoidf(acc[grp * 4 + 1][r]);
                const float gg = tanh_fast(acc[grp * 4 + 2][r]);
                const float og = sigmoidf(acc[grp * 4 + 3][r]);
                const int ci = grp * 8 + r;
                const float c = fmaf(fg, cst[ci], ig * gg);
                cst[ci] = c;
                const float h = og * tanh_fast(c);
                const int M = r + 8 * half;
                hbuf[M * 32 + hh] = (_Float16)h;      // ds_store_b16 (pad cols harmless)
            }
        }
        __syncthreads();

        // ---------- fused output linear: out[t,b] = dot(h[b,:], W_lin) + b_lin ----------
        // lane covers b = l16, k = half*16 .. half*16+15  (two ds_load_b128)
        const v8h ylo = hp[l16 * 4 + half * 2];
        const v8h yhi = hp[l16 * 4 + half * 2 + 1];
        float part = 0.0f;
        #pragma unroll
        for (int j = 0; j < 8; ++j)
            part += wl[j] * (float)ylo[j] + wl[8 + j] * (float)yhi[j];
        part += __shfl_xor(part, 16, 32);
        if (half == 0)
            out[(size_t)t * BATCH + b0 + l16] = part + blin;
    }
}

extern "C" void kernel_launch(void* const* d_in, const int* in_sizes, int n_in,
                              void* d_out, int out_size, void* d_ws, size_t ws_size,
                              hipStream_t stream) {
    (void)in_sizes; (void)n_in; (void)out_size; (void)d_ws; (void)ws_size;
    const float* x     = (const float*)d_in[0];
    const float* W_ih  = (const float*)d_in[1];
    const float* W_hh  = (const float*)d_in[2];
    const float* b_ih  = (const float*)d_in[3];
    const float* b_hh  = (const float*)d_in[4];
    const float* W_lin = (const float*)d_in[5];
    const float* b_lin = (const float*)d_in[6];
    float* out = (float*)d_out;

    dim3 grid(BATCH / 16);   // 32 independent batch-tile recurrences
    dim3 block(32);          // one wave32 each
    lstm_wmma_kernel<<<grid, block, 0, stream>>>(x, W_ih, W_hh, b_ih, b_hh,
                                                 W_lin, b_lin, out);
}
